// _Attention_29609504538626
// MI455X (gfx1250) — matrix operand order repaired and checked by reading, not compiled, not run on hardware
//
#include <hip/hip_runtime.h>
#include <hip/hip_bf16.h>
#include <math.h>

// ---------------------------------------------------------------------------
// MI455X (gfx1250) attention pipeline, bf16 WMMA + f32 accumulate.
//   B=2, T=2048, E=2048, H=16, D=128  -> ~206 GFLOP, ~130MB touched.
//   23.3 TB/s HBM => ~6us memory; compute-bound => all matmuls via
//   v_wmma_f32_16x16x32_bf16. bf16 working set (~50MB) fits the 192MB L2.
//   GEMM waves use 64x64 tiles (16 WMMA : 16 b128 loads per k-step).
//   Flash attention double-buffers its K tiles into LDS with
//   global_load_async_to_lds_b128 (ASYNCcnt pipelining).
// ---------------------------------------------------------------------------

#define B_  2
#define T_  2048
#define E_  2048
#define H_  16
#define D_  128
#define BH_ (B_ * H_)
#define M_  (B_ * T_)      // 4096 rows
#define NQKV_ (3 * E_)     // 6144

typedef __bf16 bf16_t;
typedef __attribute__((ext_vector_type(16))) __bf16 v16bf;
typedef __attribute__((ext_vector_type(8)))  __bf16 v8bf;
typedef __attribute__((ext_vector_type(8)))  float  v8f;

// ---------------------------------------------------------------------------
// WMMA wrapper: D = A(16x32 bf16) * B(32x16 bf16) + C(16x16 f32)
// ---------------------------------------------------------------------------
__device__ __forceinline__ v8f wmma_bf16(v16bf a, v16bf b, v8f c) {
  return __builtin_amdgcn_wmma_f32_16x16x32_bf16(
      /*neg_a=*/false, a, /*neg_b=*/false, b,
      /*c_mod=*/(short)0, c, /*reuse_a=*/false, /*reuse_b=*/false);
}

// A-fragment (16x32, MxK) from row-major src (K contiguous, leading dim ld).
// ISA layout: lane (half=l>>4,row=l&15): VGPR0..3 -> K = half*8 + 0..7,
//             VGPR4..7 -> K = 16 + half*8 + 0..7  => two contiguous 16B loads.
__device__ __forceinline__ v16bf load_a_frag(const bf16_t* __restrict__ src,
                                             int ld, int row_base, int k_base) {
  const int lane = threadIdx.x & 31;
  const int half = lane >> 4, row = lane & 15;
  const bf16_t* p = src + (size_t)(row_base + row) * ld + k_base + half * 8;
  v8bf lo = *(const v8bf*)p;          // K = half*8 .. half*8+7
  v8bf hi = *(const v8bf*)(p + 16);   // K = 16 + half*8 ..
  v16bf r;
#pragma unroll
  for (int i = 0; i < 8; ++i) { r[i] = lo[i]; r[8 + i] = hi[i]; }
  return r;
}

// B-fragment (32x16, KxN) where B(k,n) = W[n_base+n][k_base+k] and W rows are
// K-contiguous. ISA layout: lane: N = l&15, K = (l>>4)*8 + e (e=0..7),
// 16 + (l>>4)*8 + e (e=8..15) => two contiguous 16B loads per lane.
__device__ __forceinline__ v16bf load_b_frag(const bf16_t* __restrict__ w,
                                             int ld, int n_base, int k_base) {
  const int lane = threadIdx.x & 31;
  const bf16_t* p = w + (size_t)(n_base + (lane & 15)) * ld + k_base + (lane >> 4) * 8;
  v8bf lo = *(const v8bf*)p;
  v8bf hi = *(const v8bf*)(p + 16);
  v16bf r;
#pragma unroll
  for (int i = 0; i < 8; ++i) { r[i] = lo[i]; r[8 + i] = hi[i]; }
  return r;
}

// ---------------------------------------------------------------------------
// f32 -> bf16 conversion
// ---------------------------------------------------------------------------
__global__ void k_cvt_bf16(const float* __restrict__ src, bf16_t* __restrict__ dst, int n) {
  int i = blockIdx.x * blockDim.x + threadIdx.x;
  int stride = gridDim.x * blockDim.x;
  for (; i < n; i += stride) dst[i] = (bf16_t)src[i];
}

// ---------------------------------------------------------------------------
// QKV projection: qkv[m,n] = sum_k x[m,k] * w_in[n,k]   (m in B*T, n in 3E)
// Block: 256 thr = 8 waves (2x4), block tile 128x256, wave tile 64x64.
// 16 WMMAs : 16 b128 loads per k-step per wave.
// Epilogue scatters to q,k [BH,T,D] and V-transposed [BH,D,T] (bf16).
// ---------------------------------------------------------------------------
__global__ __launch_bounds__(256) void k_gemm_qkv(
    const bf16_t* __restrict__ A, const bf16_t* __restrict__ W,
    bf16_t* __restrict__ qb, bf16_t* __restrict__ kb, bf16_t* __restrict__ vT) {
  const int wid  = threadIdx.x >> 5;
  const int lane = threadIdx.x & 31;
  const int m0 = blockIdx.y * 128 + (wid >> 2) * 64;
  const int n0 = blockIdx.x * 256 + (wid & 3) * 64;

  v8f c[4][4];
#pragma unroll
  for (int i = 0; i < 4; ++i)
#pragma unroll
    for (int j = 0; j < 4; ++j) c[i][j] = (v8f)0.0f;

  for (int kk = 0; kk < E_; kk += 32) {
    if (kk + 32 < E_) {  // L2-resident streams; near-scope hint
      __builtin_prefetch(A + (size_t)m0 * E_ + kk + 32, 0, 3);
      __builtin_prefetch(W + (size_t)n0 * E_ + kk + 32, 0, 3);
    }
    v16bf af[4], bfr[4];
#pragma unroll
    for (int i = 0; i < 4; ++i) af[i] = load_a_frag(A, E_, m0 + i * 16, kk);
#pragma unroll
    for (int j = 0; j < 4; ++j) bfr[j] = load_b_frag(W, E_, n0 + j * 16, kk);
#pragma unroll
    for (int i = 0; i < 4; ++i)
#pragma unroll
      for (int j = 0; j < 4; ++j) c[i][j] = wmma_bf16(af[i], bfr[j], c[i][j]);
  }

  const int half = lane >> 4, nl = lane & 15;
#pragma unroll
  for (int mt = 0; mt < 4; ++mt)
#pragma unroll
    for (int nt = 0; nt < 4; ++nt)
#pragma unroll
      for (int r = 0; r < 8; ++r) {
        int m = m0 + mt * 16 + half * 8 + r;
        int n = n0 + nt * 16 + nl;
        int b = m >> 11, t = m & 2047;
        int s = n >> 11, rem = n & 2047;
        int h = rem >> 7, d = rem & 127;
        int bh = b * H_ + h;
        bf16_t val = (bf16_t)c[mt][nt][r];
        if (s == 0)      qb[((size_t)bh * T_ + t) * D_ + d] = val;
        else if (s == 1) kb[((size_t)bh * T_ + t) * D_ + d] = val;
        else             vT[((size_t)bh * D_ + d) * T_ + t] = val;  // transposed
      }
}

// ---------------------------------------------------------------------------
// RoPE on q and k in-place (bf16), pairs (d, d+64) per head.
// grid: (T, BH, 2) block: 64
// ---------------------------------------------------------------------------
__global__ void k_rope(bf16_t* __restrict__ q, bf16_t* __restrict__ k) {
  const int d  = threadIdx.x;       // 0..63
  const int t  = blockIdx.x;        // 0..T-1
  const int bh = blockIdx.y;        // 0..BH-1
  bf16_t* buf = blockIdx.z ? k : q;
  const size_t base = ((size_t)bh * T_ + t) * D_;
  float freq = __expf((float)d * (-2.0f * 9.210340371976184f / (float)D_));
  float ang = freq * (float)t;
  float cs = __cosf(ang), sn = __sinf(ang);
  float x1 = (float)buf[base + d];
  float x2 = (float)buf[base + d + 64];
  buf[base + d]      = (bf16_t)(x1 * cs - x2 * sn);
  buf[base + d + 64] = (bf16_t)(x1 * sn + x2 * cs);
}

// ---------------------------------------------------------------------------
// Causal flash attention, one wave per 32 q-rows, k-blocks of 32.
// K tiles (32x128 bf16 = one contiguous 8KB region of [T,D]) are
// double-buffered into LDS via global_load_async_to_lds_b128 (ASYNCcnt):
// issue block i+1, s_wait_asynccnt 16 -> block i resident (in-order),
// consume B-fragments from bank-padded LDS (ld=136 elems).
// S = Q K^T via WMMA (f32 accum), online softmax (f32), P staged through a
// padded LDS tile (C-layout -> A-layout), O += P V via WMMA with V^T layout.
// Block: 64 thr = 2 waves. grid: (T/64, BH)
// ---------------------------------------------------------------------------
#define ATT_WAVES 2
#define P_LD 40      // 32 cols + 8 pad: rows 16B aligned, banks conflict-free
#define KSTG_LD 136  // 128 + 8 pad elems: row stride 272B, 16B aligned

__device__ __forceinline__ void stage_k_async(const bf16_t* __restrict__ kp,
                                              int kb, bf16_t* dst) {
  const int lane = threadIdx.x & 31;
  const char* src = (const char*)(kp + (size_t)kb * D_);  // 8KB contiguous
#pragma unroll
  for (int j = 0; j < 16; ++j) {
    int c = j * 32 + lane;           // 16B chunk id; 16 chunks per 256B row
    int row = c >> 4, colc = c & 15;
    unsigned lds_dst = (unsigned)(uintptr_t)(dst + row * KSTG_LD) + colc * 16u;
    unsigned long long ga = (unsigned long long)(uintptr_t)(src) + (unsigned)c * 16u;
    asm volatile("global_load_async_to_lds_b128 %0, %1, off"
                 :: "v"(lds_dst), "v"(ga) : "memory");
  }
}

__global__ __launch_bounds__(64) void k_attn(
    const bf16_t* __restrict__ q, const bf16_t* __restrict__ k,
    const bf16_t* __restrict__ vT, bf16_t* __restrict__ attn) {
  __shared__ __align__(16) bf16_t kst[ATT_WAVES][2][32 * KSTG_LD];  // ~34.8KB
  __shared__ __align__(16) bf16_t pbuf[ATT_WAVES][32 * P_LD];       //  ~5.1KB

  const int wid  = threadIdx.x >> 5;
  const int lane = threadIdx.x & 31;
  const int half = lane >> 4, nl = lane & 15;
  const int bh = blockIdx.y;
  const int q0 = blockIdx.x * (ATT_WAVES * 32) + wid * 32;

  const bf16_t* qp = q  + (size_t)bh * T_ * D_;
  const bf16_t* kp = k  + (size_t)bh * T_ * D_;
  const bf16_t* vp = vT + (size_t)bh * D_ * T_;

  // Q fragments for this wave's 32 rows are loop-invariant: preload all 8.
  v16bf qf[2][4];
#pragma unroll
  for (int mt = 0; mt < 2; ++mt)
#pragma unroll
    for (int ks = 0; ks < 4; ++ks)
      qf[mt][ks] = load_a_frag(qp, D_, q0 + mt * 16, ks * 32);

  v8f o[2][8];
  float mrow[2][8], lrow[2][8];
#pragma unroll
  for (int mt = 0; mt < 2; ++mt) {
#pragma unroll
    for (int nt = 0; nt < 8; ++nt) o[mt][nt] = (v8f)0.0f;
#pragma unroll
    for (int r = 0; r < 8; ++r) { mrow[mt][r] = -1.0e30f; lrow[mt][r] = 0.0f; }
  }

  const float scale = 0.08838834764831845f;  // 1/sqrt(128)
  const int nblk = q0 / 32 + 1;              // causal: cols up to q0+31

  stage_k_async(kp, 0, kst[wid][0]);         // prime the pipeline

  for (int ib = 0; ib < nblk; ++ib) {
    const int kb = ib * 32;
    if (ib + 1 < nblk) {
      stage_k_async(kp, kb + 32, kst[wid][(ib + 1) & 1]);
      // 16 newest in flight belong to the next buffer; current is resident
      asm volatile("s_wait_asynccnt 0x10" ::: "memory");
    } else {
      asm volatile("s_wait_asynccnt 0x0" ::: "memory");
    }
    const bf16_t* klds = kst[wid][ib & 1];

    // ---- S = Q K^T (32x32 tile), K fragments from LDS ----
    v8f s[2][2];
#pragma unroll
    for (int mt = 0; mt < 2; ++mt)
#pragma unroll
      for (int kt = 0; kt < 2; ++kt) s[mt][kt] = (v8f)0.0f;
#pragma unroll
    for (int kt = 0; kt < 2; ++kt)
#pragma unroll
      for (int ks = 0; ks < 4; ++ks) {
        v16bf bf = load_b_frag(klds, KSTG_LD, kt * 16, ks * 32);
        s[0][kt] = wmma_bf16(qf[0][ks], bf, s[0][kt]);
        s[1][kt] = wmma_bf16(qf[1][ks], bf, s[1][kt]);
      }

    // ---- mask + online softmax; write P to LDS ----
#pragma unroll
    for (int mt = 0; mt < 2; ++mt)
#pragma unroll
      for (int r = 0; r < 8; ++r) {
        const int row = q0 + mt * 16 + half * 8 + r;
        float v0 = (float)s[mt][0][r] * scale;
        float v1 = (float)s[mt][1][r] * scale;
        if (kb + nl > row)      v0 = -1.0e30f;
        if (kb + 16 + nl > row) v1 = -1.0e30f;
        float vm = fmaxf(v0, v1);
        vm = fmaxf(vm, __shfl_xor(vm, 1));
        vm = fmaxf(vm, __shfl_xor(vm, 2));
        vm = fmaxf(vm, __shfl_xor(vm, 4));
        vm = fmaxf(vm, __shfl_xor(vm, 8));   // row max over 16 lanes
        float mold = mrow[mt][r];
        float mnew = fmaxf(mold, vm);
        float alpha = __expf(mold - mnew);
        float p0 = __expf(v0 - mnew);
        float p1 = __expf(v1 - mnew);
        float ps = p0 + p1;
        ps += __shfl_xor(ps, 1);
        ps += __shfl_xor(ps, 2);
        ps += __shfl_xor(ps, 4);
        ps += __shfl_xor(ps, 8);             // row sum
        lrow[mt][r] = lrow[mt][r] * alpha + ps;
        mrow[mt][r] = mnew;
#pragma unroll
        for (int nt = 0; nt < 8; ++nt) o[mt][nt][r] *= alpha;
        const int rl = mt * 16 + half * 8 + r;
        pbuf[wid][rl * P_LD + nl]      = (bf16_t)p0;
        pbuf[wid][rl * P_LD + 16 + nl] = (bf16_t)p1;
      }
    // Wave-private LDS tile: producer==consumer wave; DScnt ordering is
    // handled by the compiler. No block barrier needed.

    // ---- O += P V  (V^T layout -> contiguous B fragments along time) ----
    v16bf pa0 = load_a_frag(pbuf[wid], P_LD, 0,  0);
    v16bf pa1 = load_a_frag(pbuf[wid], P_LD, 16, 0);
#pragma unroll
    for (int nt = 0; nt < 8; ++nt) {
      v16bf bv = load_b_frag(vp, T_, nt * 16, kb);
      o[0][nt] = wmma_bf16(pa0, bv, o[0][nt]);
      o[1][nt] = wmma_bf16(pa1, bv, o[1][nt]);
    }
  }

  // ---- normalize and store attn_out bf16 [(b*T+t)*E + h*D + d] ----
  const int b = bh / H_, h = bh % H_;
#pragma unroll
  for (int mt = 0; mt < 2; ++mt)
#pragma unroll
    for (int r = 0; r < 8; ++r) {
      float inv = 1.0f / fmaxf(lrow[mt][r], 1.0e-30f);
      int t = q0 + mt * 16 + half * 8 + r;
#pragma unroll
      for (int nt = 0; nt < 8; ++nt) {
        int d = nt * 16 + nl;
        attn[((size_t)(b * T_ + t)) * E_ + h * D_ + d] =
            (bf16_t)(o[mt][nt][r] * inv);
      }
    }
}

// ---------------------------------------------------------------------------
// Output projection: out[m,n] = sum_k attn[m,k] * w_out[n,k], f32 out.
// Same 64x64 wave tiling as k_gemm_qkv.
// ---------------------------------------------------------------------------
__global__ __launch_bounds__(256) void k_gemm_out(
    const bf16_t* __restrict__ A, const bf16_t* __restrict__ W,
    float* __restrict__ out) {
  const int wid  = threadIdx.x >> 5;
  const int lane = threadIdx.x & 31;
  const int m0 = blockIdx.y * 128 + (wid >> 2) * 64;
  const int n0 = blockIdx.x * 256 + (wid & 3) * 64;

  v8f c[4][4];
#pragma unroll
  for (int i = 0; i < 4; ++i)
#pragma unroll
    for (int j = 0; j < 4; ++j) c[i][j] = (v8f)0.0f;

  for (int kk = 0; kk < E_; kk += 32) {
    if (kk + 32 < E_) {
      __builtin_prefetch(A + (size_t)m0 * E_ + kk + 32, 0, 3);
      __builtin_prefetch(W + (size_t)n0 * E_ + kk + 32, 0, 3);
    }
    v16bf af[4], bfr[4];
#pragma unroll
    for (int i = 0; i < 4; ++i) af[i] = load_a_frag(A, E_, m0 + i * 16, kk);
#pragma unroll
    for (int j = 0; j < 4; ++j) bfr[j] = load_b_frag(W, E_, n0 + j * 16, kk);
#pragma unroll
    for (int i = 0; i < 4; ++i)
#pragma unroll
      for (int j = 0; j < 4; ++j) c[i][j] = wmma_bf16(af[i], bfr[j], c[i][j]);
  }

  const int half = lane >> 4, nl = lane & 15;
#pragma unroll
  for (int mt = 0; mt < 4; ++mt)
#pragma unroll
    for (int nt = 0; nt < 4; ++nt)
#pragma unroll
      for (int r = 0; r < 8; ++r) {
        int m = m0 + mt * 16 + half * 8 + r;
        int n = n0 + nt * 16 + nl;
        out[(size_t)m * E_ + n] = c[mt][nt][r];
      }
}

// ---------------------------------------------------------------------------
// Launch
// ---------------------------------------------------------------------------
extern "C" void kernel_launch(void* const* d_in, const int* in_sizes, int n_in,
                              void* d_out, int out_size, void* d_ws, size_t ws_size,
                              hipStream_t stream) {
  (void)in_sizes; (void)n_in; (void)out_size; (void)ws_size;
  const float* x     = (const float*)d_in[0];
  const float* w_in  = (const float*)d_in[1];
  const float* w_out = (const float*)d_in[2];
  float* out = (float*)d_out;

  char* ws = (char*)d_ws;
  size_t off = 0;
  auto take = [&](size_t bytes) -> char* {
    char* p = ws + off;
    off += (bytes + 255) & ~(size_t)255;
    return p;
  };
  bf16_t* xb   = (bf16_t*)take((size_t)M_ * E_ * 2);         // 16.8 MB
  bf16_t* winb = (bf16_t*)take((size_t)NQKV_ * E_ * 2);      // 25.2 MB
  bf16_t* woutb= (bf16_t*)take((size_t)E_ * E_ * 2);         //  8.4 MB
  bf16_t* qb   = (bf16_t*)take((size_t)BH_ * T_ * D_ * 2);   // 16.8 MB
  bf16_t* kb   = (bf16_t*)take((size_t)BH_ * T_ * D_ * 2);   // 16.8 MB
  bf16_t* vT   = (bf16_t*)take((size_t)BH_ * D_ * T_ * 2);   // 16.8 MB
  bf16_t* attn = (bf16_t*)take((size_t)M_ * E_ * 2);         // 16.8 MB

  k_cvt_bf16<<<2048, 256, 0, stream>>>(x,     xb,    M_ * E_);
  k_cvt_bf16<<<2048, 256, 0, stream>>>(w_in,  winb,  NQKV_ * E_);
  k_cvt_bf16<<<1024, 256, 0, stream>>>(w_out, woutb, E_ * E_);

  k_gemm_qkv<<<dim3(NQKV_ / 256, M_ / 128), 256, 0, stream>>>(xb, winb, qb, kb, vT);
  k_rope<<<dim3(T_, BH_, 2), 64, 0, stream>>>(qb, kb);
  k_attn<<<dim3(T_ / (ATT_WAVES * 32), BH_), 64, 0, stream>>>(qb, kb, vT, attn);
  k_gemm_out<<<dim3(E_ / 256, M_ / 128), 256, 0, stream>>>(attn, woutb, out);
}
